// DeepFeatureLoss_979252544039
// MI455X (gfx1250) — matrix operand order repaired and checked by reading, not compiled, not run on hardware
//
#include <hip/hip_runtime.h>

#define SIGMA_INV 20.0f      // 1/0.05
#define D_FIX 64
#define JT 128               // j-chunk (columns) staged in LDS per iteration
#define BLK 256              // 8 waves (wave32)

typedef _Float16 v16h __attribute__((ext_vector_type(16)));
typedef float    v8f  __attribute__((ext_vector_type(8)));
typedef float    v2f  __attribute__((ext_vector_type(2)));
typedef int      v4i  __attribute__((vector_size(16)));   // matches async-LDS builtin param

typedef __attribute__((address_space(1))) v4i* g_v4i_ptr;  // global
typedef __attribute__((address_space(3))) v4i* l_v4i_ptr;  // LDS

#if __has_builtin(__builtin_amdgcn_global_load_async_to_lds_b128) && __has_builtin(__builtin_amdgcn_s_wait_asynccnt)
#define USE_ASYNC 1
#else
#define USE_ASYNC 0
#endif

#if __has_builtin(__builtin_amdgcn_wmma_f32_16x16x4_f32)
#define USE_WMMA_SPATIAL 1
#else
#define USE_WMMA_SPATIAL 0
#endif

union U16H { _Float16 h[16]; uint4 q[2]; v16h v; };

__global__ void dfl_zero(float* out, int n) {
  if ((int)threadIdx.x < n) out[threadIdx.x] = 0.f;
}

__global__ __launch_bounds__(BLK) void dfl_main(const float* __restrict__ points,
                                                const float* __restrict__ weights,
                                                const float* __restrict__ fea1,
                                                const float* __restrict__ fea2,
                                                float* __restrict__ out,
                                                int Nn) {
  const int b      = blockIdx.y;
  const int iblock = blockIdx.x * 128;          // 8 waves * 16 rows
  const int tid    = threadIdx.x;
  const int wave   = tid >> 5;
  const int lane   = tid & 31;
  const int sel    = lane >> 4;                 // half-wave select
  const int lcol   = lane & 15;

#if USE_ASYNC
  __shared__ __align__(16) float stag[JT * D_FIX];        // 32 KB f32 staging
#endif
  __shared__ __align__(16) unsigned short f2h[JT][D_FIX]; // 16 KB f16 B-tiles
  __shared__ float fnpart[JT][2];
  __shared__ float fn[JT];                                // |f2_j|^2
  __shared__ __align__(16) float xsj[JT][4];              // scaled x_j (pad to 4)
  __shared__ float xn[JT];                                // |x_j/sigma|^2
  __shared__ __align__(16) float xsi[128][4];             // scaled x_i rows (VALU fallback)

  // ---- per-wave A matrices (fixed for whole kernel) ----
  const int rowA = iblock + wave * 16 + lcol;   // this lane's A row
  U16H aF0, aF1;                                // feature A, premultiplied by 2
  {
    const float* f1r = fea1 + ((size_t)b * Nn + rowA) * D_FIX;
    const int k0 = sel * 8;
#pragma unroll
    for (int h = 0; h < 8; ++h) {
      aF0.h[h]     = (_Float16)(2.0f * f1r[k0 + h]);
      aF0.h[8 + h] = (_Float16)(2.0f * f1r[k0 + 16 + h]);
      aF1.h[h]     = (_Float16)(2.0f * f1r[32 + k0 + h]);
      aF1.h[8 + h] = (_Float16)(2.0f * f1r[32 + k0 + 16 + h]);
    }
  }
#if USE_WMMA_SPATIAL
  v2f aS;                                       // spatial A = 2*x_i/sigma
  {
    const float* pr = points + ((size_t)b * Nn + rowA) * 3;
    const int c0 = sel * 2;                     // 0 or 2
    aS.x = 2.0f * SIGMA_INV * pr[c0];
    aS.y = (c0 + 1 < 3) ? 2.0f * SIGMA_INV * pr[c0 + 1] : 0.0f;
  }
#endif
  if (tid < 128) {
    const float* pr = points + ((size_t)b * Nn + iblock + tid) * 3;
    xsi[tid][0] = pr[0] * SIGMA_INV;
    xsi[tid][1] = pr[1] * SIGMA_INV;
    xsi[tid][2] = pr[2] * SIGMA_INV;
    xsi[tid][3] = 0.f;
  }

  // |xs_i|^2 for this lane's 8 C-rows -> spatial logits become -||xs_i-xs_j||^2
  // (exact row max 0 at j=i: no online max tracking needed for the spatial softmax)
  float xin[8];
#pragma unroll
  for (int r = 0; r < 8; ++r) {
    const float* pr = points + ((size_t)b * Nn + iblock + wave * 16 + sel * 8 + r) * 3;
    float x = pr[0] * SIGMA_INV, y = pr[1] * SIGMA_INV, z = pr[2] * SIGMA_INV;
    xin[r] = x * x + y * y + z * z;
  }

  // ---- accumulator state: per lane, 8 C-rows each ----
  float la[8], saa[8];                 // spatial: sum e, sum e^2 (ref max = 0)
  float mb[8], lb[8], sbb[8];          // feature: running max, sum e, sum e^2
  float sab[8];                        // cross term sum ea*eb
#pragma unroll
  for (int r = 0; r < 8; ++r) {
    la[r] = 0.f; saa[r] = 0.f;
    mb[r] = -1e30f; lb[r] = 0.f; sbb[r] = 0.f; sab[r] = 0.f;
  }

  for (int j0 = 0; j0 < Nn; j0 += JT) {
    __syncthreads();                            // previous chunk fully consumed
    // ---- stage f2 chunk ----
#if USE_ASYNC
    {
      char* gbase = (char*)(fea2 + ((size_t)b * Nn + j0) * D_FIX);
      char* lbase = (char*)&stag[0];
#pragma unroll
      for (int q = 0; q < 8; ++q) {
        __builtin_amdgcn_global_load_async_to_lds_b128(
            (g_v4i_ptr)(void*)(gbase + tid * 16 + q * 4096),
            (l_v4i_ptr)(void*)(lbase + tid * 16 + q * 4096),
            0, 0);
      }
    }
#endif
    if (tid < JT) {
      const float* pr = points + ((size_t)b * Nn + j0 + tid) * 3;
      float x = pr[0] * SIGMA_INV, y = pr[1] * SIGMA_INV, z = pr[2] * SIGMA_INV;
      xsj[tid][0] = x; xsj[tid][1] = y; xsj[tid][2] = z; xsj[tid][3] = 0.f;
      xn[tid] = x * x + y * y + z * z;
    }
#if USE_ASYNC
    __builtin_amdgcn_s_wait_asynccnt(0);
#endif
    __syncthreads();
    // ---- convert f32 -> f16 into WMMA B-friendly layout + partial norms ----
    {
      const int j = tid >> 1, half = tid & 1;
#if USE_ASYNC
      const float* srcp = &stag[j * D_FIX + half * 32];
#else
      const float* srcp = fea2 + ((size_t)b * Nn + j0 + j) * D_FIX + half * 32;
#endif
      unsigned short* dst = &f2h[j][half * 32];
      float ss = 0.f;
#pragma unroll
      for (int q = 0; q < 8; ++q) {
        float4 v = *(const float4*)(srcp + q * 4);
        ss = fmaf(v.x, v.x, fmaf(v.y, v.y, fmaf(v.z, v.z, fmaf(v.w, v.w, ss))));
        union { _Float16 h[4]; uint2 u2; } pk;
        pk.h[0] = (_Float16)v.x; pk.h[1] = (_Float16)v.y;
        pk.h[2] = (_Float16)v.z; pk.h[3] = (_Float16)v.w;
        *(uint2*)(dst + q * 4) = pk.u2;
      }
      fnpart[j][half] = ss;
    }
    __syncthreads();
    if (tid < JT) fn[tid] = fnpart[tid][0] + fnpart[tid][1];
    __syncthreads();

    if (j0 + JT < Nn)
      __builtin_prefetch((const char*)(fea2 + ((size_t)b * Nn + j0 + JT) * D_FIX) + tid * 128, 0, 3);

    // ---- pass A: chunk-local feature-logit max (WMMA + fmax only, no exps) ----
    float mloc[8];
#pragma unroll
    for (int r = 0; r < 8; ++r) mloc[r] = -1e30f;
#pragma unroll 1
    for (int jt = 0; jt < JT; jt += 16) {
      const int col = jt + lcol;
      U16H b0, b1;
      const uint4* rowp = (const uint4*)&f2h[col][0];
      b0.q[0] = rowp[sel + 0];
      b0.q[1] = rowp[sel + 2];
      b1.q[0] = rowp[4 + sel + 0];
      b1.q[1] = rowp[4 + sel + 2];
      v8f cz = {};
      v8f cF = __builtin_amdgcn_wmma_f32_16x16x32_f16(false, aF0.v, false, b0.v,
                                                      (short)0, cz, false, false);
      cF = __builtin_amdgcn_wmma_f32_16x16x32_f16(false, aF1.v, false, b1.v,
                                                  (short)0, cF, false, false);
      const float fnj = fn[col];
#pragma unroll
      for (int r = 0; r < 8; ++r) mloc[r] = fmaxf(mloc[r], cF[r] - fnj);
    }
    // one rescale per chunk per row (amortized over 128 columns)
#pragma unroll
    for (int r = 0; r < 8; ++r) {
      float mu = fmaxf(mb[r], mloc[r]);
      float s  = __expf(mb[r] - mu);
      lb[r] *= s; sbb[r] *= s * s; sab[r] *= s;
      mb[r] = mu;
    }

    // ---- pass B: accumulate with exactly 2 exps per element ----
#pragma unroll 1
    for (int jt = 0; jt < JT; jt += 16) {
      const int col = jt + lcol;
      U16H b0, b1;
      const uint4* rowp = (const uint4*)&f2h[col][0];
      b0.q[0] = rowp[sel + 0];
      b0.q[1] = rowp[sel + 2];
      b1.q[0] = rowp[4 + sel + 0];
      b1.q[1] = rowp[4 + sel + 2];
      v8f cz = {};
      v8f cF = __builtin_amdgcn_wmma_f32_16x16x32_f16(false, aF0.v, false, b0.v,
                                                      (short)0, cz, false, false);
      cF = __builtin_amdgcn_wmma_f32_16x16x32_f16(false, aF1.v, false, b1.v,
                                                  (short)0, cF, false, false);
      float sdot[8];
#if USE_WMMA_SPATIAL
      {
        v2f bS;
        bS.x = xsj[col][sel * 2 + 0];
        bS.y = xsj[col][sel * 2 + 1];
        v8f cS = __builtin_amdgcn_wmma_f32_16x16x4_f32(false, aS, false, bS,
                                                       (short)0, cz, false, false);
#pragma unroll
        for (int r = 0; r < 8; ++r) sdot[r] = cS[r];
      }
#else
      {
        const float bx = xsj[col][0], by = xsj[col][1], bz = xsj[col][2];
#pragma unroll
        for (int r = 0; r < 8; ++r) {
          const float* xr = &xsi[wave * 16 + sel * 8 + r][0];
          sdot[r] = 2.0f * (xr[0] * bx + xr[1] * by + xr[2] * bz);
        }
      }
#endif
      const float xnj = xn[col], fnj = fn[col];
#pragma unroll
      for (int r = 0; r < 8; ++r) {
        float ea = __expf(sdot[r] - xnj - xin[r]);   // = exp(-||xs_i-xs_j||^2) <= 1
        float eb = __expf(cF[r] - fnj - mb[r]);      // <= 1 (mb >= chunk max)
        la[r]  += ea;
        saa[r]  = fmaf(ea, ea, saa[r]);
        lb[r]  += eb;
        sbb[r]  = fmaf(eb, eb, sbb[r]);
        sab[r]  = fmaf(ea, eb, sab[r]);
      }
    }
  }

  // ---- merge the 16 lanes sharing each C-row (butterfly within half-wave) ----
#pragma unroll
  for (int r = 0; r < 8; ++r) {
#pragma unroll
    for (int off = 1; off < 16; off <<= 1) {
      float la2  = __shfl_xor(la[r],  off, 32);
      float sa2  = __shfl_xor(saa[r], off, 32);
      float mb2  = __shfl_xor(mb[r],  off, 32);
      float lb2  = __shfl_xor(lb[r],  off, 32);
      float sb2  = __shfl_xor(sbb[r], off, 32);
      float sx2  = __shfl_xor(sab[r], off, 32);
      la[r]  += la2;                       // spatial: absolute reference (max = 0)
      saa[r] += sa2;
      float Mb = fmaxf(mb[r], mb2);
      float g1 = __expf(mb[r] - Mb), g2 = __expf(mb2 - Mb);
      lb[r]  = lb[r]  * g1      + lb2 * g2;
      sbb[r] = sbb[r] * g1 * g1 + sb2 * g2 * g2;
      sab[r] = sab[r] * g1      + sx2 * g2;
      mb[r] = Mb;
    }
  }
  if (lcol == 0) {
    float acc = 0.f;
#pragma unroll
    for (int r = 0; r < 8; ++r) {
      const int i = iblock + wave * 16 + sel * 8 + r;
      float inva = 1.0f / la[r], invb = 1.0f / lb[r];
      float v = saa[r] * inva * inva + sbb[r] * invb * invb
              - 2.0f * sab[r] * inva * invb;     // sum_j (P_ij - Q_ij)^2
      acc += v * weights[(size_t)b * Nn + i];
    }
    atomicAdd(&out[b], acc);
  }
}

extern "C" void kernel_launch(void* const* d_in, const int* in_sizes, int n_in,
                              void* d_out, int out_size, void* d_ws, size_t ws_size,
                              hipStream_t stream) {
  const float* points  = (const float*)d_in[0];
  const float* weights = (const float*)d_in[1];
  const float* fea1    = (const float*)d_in[2];
  const float* fea2    = (const float*)d_in[3];
  float* out = (float*)d_out;
  const int N = 4096;
  const int B = in_sizes[1] / N;   // weights is [B,N]
  dfl_zero<<<1, 32, 0, stream>>>(out, out_size);
  dim3 grid(N / 128, B, 1);
  dfl_main<<<grid, BLK, 0, stream>>>(points, weights, fea1, fea2, out, N);
}
